// Head_71897752535044
// MI455X (gfx1250) — matrix operand order repaired and checked by reading, not compiled, not run on hardware
//
#include <hip/hip_runtime.h>

#define EMBED  256
#define HEADD  64
#define SEQT   2048
#define NBATCH 16
#define MROWS  (NBATCH * SEQT)   // 32768 total rows across batch

typedef __attribute__((ext_vector_type(16))) __bf16       v16bf;
typedef __attribute__((ext_vector_type(8)))  float        v8f;
typedef __attribute__((ext_vector_type(4)))  unsigned int v4u;
typedef __attribute__((ext_vector_type(4)))  float        v4f;

union Frag {
  v16bf          bf;
  v4u            q[2];
  unsigned short u[16];
};

// Native f32 -> bf16 convert (lowers to v_cvt_*bf16_f32, RNE) instead of a
// 3-instruction manual rounding sequence per element.
__device__ inline unsigned short f2bf(float f) {
  return __builtin_bit_cast(unsigned short, (__bf16)f);
}

// ---------------------------------------------------------------------------
// Phase 1: fused QKV projection.  [32768,256]f32 x [256,64]f32 -> [32768,64]bf16
// grid.z selects q/k/v.  Memory-bound: streams the 100 MB of activations once.
// ---------------------------------------------------------------------------
__global__ __launch_bounds__(256) void qkv_proj(
    const float* __restrict__ q, const float* __restrict__ k, const float* __restrict__ v,
    const float* __restrict__ Wq, const float* __restrict__ Wk, const float* __restrict__ Wv,
    unsigned short* __restrict__ qh, unsigned short* __restrict__ kh,
    unsigned short* __restrict__ vh) {
  const int bz = blockIdx.z;
  const float* X = (bz == 0) ? q : (bz == 1) ? k : v;
  const float* W = (bz == 0) ? Wq : (bz == 1) ? Wk : Wv;
  unsigned short* Y = (bz == 0) ? qh : (bz == 1) ? kh : vh;

  __shared__ unsigned short sX[128 * 32];   // activation tile [row][k]  (bf16)
  __shared__ unsigned short sWt[64 * 32];   // weight tile transposed [n][k]

  const int tid  = threadIdx.x;
  const int wave = tid >> 5;
  const int lane = tid & 31;
  const int hi   = lane >> 4;
  const int n0   = lane & 15;
  const int row0 = blockIdx.x * 128;        // 128 rows per block, 16 per wave

  v8f acc[4] = {};                          // 16x64 f32 accumulator per wave

  for (int k0 = 0; k0 < EMBED; k0 += 32) {
    __syncthreads();
    {   // stage X tile 128x32 f32 -> bf16 LDS (each thread: 16 floats)
      const int r = tid >> 1, hsel = (tid & 1) * 16;
      const float* src = X + (row0 + r) * EMBED + k0 + hsel;
      unsigned short* dst = sX + r * 32 + hsel;
      #pragma unroll
      for (int i = 0; i < 16; i += 4) {
        v4f f = *(const v4f*)(src + i);
        dst[i + 0] = f2bf(f[0]); dst[i + 1] = f2bf(f[1]);
        dst[i + 2] = f2bf(f[2]); dst[i + 3] = f2bf(f[3]);
      }
    }
    {   // stage W tile 32x64, transposed into [n][k]
      const int kr = tid & 31, c0 = (tid >> 5) * 8;
      const float* src = W + (k0 + kr) * HEADD + c0;
      #pragma unroll
      for (int i = 0; i < 8; ++i) sWt[(c0 + i) * 32 + kr] = f2bf(src[i]);
    }
    __syncthreads();

    // A fragment: 16x32 bf16 (rows = wave tile)
    Frag a;
    {
      const unsigned short* base = sX + (wave * 16 + n0) * 32;
      a.q[0] = *(const v4u*)(base + 8 * hi);        // K = 8*hi + 0..7
      a.q[1] = *(const v4u*)(base + 16 + 8 * hi);   // K = 16 + 8*hi + 0..7
    }
    #pragma unroll
    for (int nt = 0; nt < 4; ++nt) {                // 4 x 16 output columns
      Frag b;
      const unsigned short* wb = sWt + (nt * 16 + n0) * 32 + 8 * hi;
      b.q[0] = *(const v4u*)(wb);
      b.q[1] = *(const v4u*)(wb + 16);
      acc[nt] = __builtin_amdgcn_wmma_f32_16x16x32_bf16(
          false, a.bf, false, b.bf, (short)0, acc[nt], false, false);
    }
  }

  #pragma unroll
  for (int nt = 0; nt < 4; ++nt)
    #pragma unroll
    for (int r = 0; r < 8; ++r) {
      const int row = row0 + wave * 16 + r + 8 * hi;   // D layout: VGPR r -> M=r / r+8
      Y[row * HEADD + nt * 16 + n0] = f2bf(acc[nt][r]);
    }
}

// ---------------------------------------------------------------------------
// Phase 2: causal flash attention over bf16 qh/kh/vh, f32 online softmax.
// Block = 8 waves = 128 query rows of one batch; K tile staged with CDNA5
// async global->LDS copies (ASYNCcnt), V tile transposed in LDS; causality
// computed inline (mask input is a tril) and fully-masked key tiles skipped.
// ---------------------------------------------------------------------------
__global__ __launch_bounds__(256) void flash_attn(
    const unsigned short* __restrict__ qh, const unsigned short* __restrict__ kh,
    const unsigned short* __restrict__ vh, float* __restrict__ out) {
  __shared__ unsigned short sK[32 * 64];      // [key][h]   row-major for Q·K^T
  __shared__ unsigned short sVt[64 * 32];     // [h][key]   transposed for P·V
  __shared__ unsigned short sP[8][16 * 32];   // per-wave P tile (D -> A relayout)

  const int tid  = threadIdx.x;
  const int wave = tid >> 5;
  const int lane = tid & 31;
  const int hi   = lane >> 4;
  const int n0   = lane & 15;
  const int b    = blockIdx.y;
  const int row0 = blockIdx.x * 128 + wave * 16;   // wave's 16-row query tile
  const int baseRow = b * SEQT;

  // LDS byte offset of sK within this workgroup's allocation (generic LDS
  // addresses carry the offset in the low 32 bits).
  const unsigned ldsK = (unsigned)(unsigned long long)(size_t)(const void*)sK;

  // Q fragments: 16x64 bf16 = two K=32 A-fragments, loaded once.
  Frag aq[2];
  {
    const unsigned short* qr = qh + (size_t)(baseRow + row0 + n0) * HEADD;
    #pragma unroll
    for (int c = 0; c < 2; ++c) {
      aq[c].q[0] = *(const v4u*)(qr + 32 * c + 8 * hi);
      aq[c].q[1] = *(const v4u*)(qr + 32 * c + 16 + 8 * hi);
    }
  }

  v8f   o[4] = {};
  float mrow[8], lrow[8];
  #pragma unroll
  for (int r = 0; r < 8; ++r) { mrow[r] = -1e30f; lrow[r] = 0.f; }

  const float scale = 0.0625f;                    // EMBED^-0.5 = 1/16 (reference uses C=E)
  const int   sEnd  = blockIdx.x * 128 + 127;     // causal bound for this block

  for (int s0 = 0; s0 <= sEnd; s0 += 32) {
    __syncthreads();
    {   // K tile: async global->LDS byte copy (32 keys x 64 dims, 16B/lane)
      const unsigned long long gaddr =
          (unsigned long long)(const void*)(kh + (size_t)(baseRow + s0) * HEADD + tid * 8);
      const unsigned dsaddr = ldsK + (unsigned)tid * 16u;
      asm volatile("global_load_async_to_lds_b128 %0, %1, off"
                   :: "v"(dsaddr), "v"(gaddr) : "memory");
    }
    {   // V tile transposed into [h][key] (layout change -> VALU/ds path)
      const int key = tid & 31, h0 = (tid >> 5) * 8;
      const unsigned short* vsrc = vh + (size_t)(baseRow + s0 + key) * HEADD + h0;
      #pragma unroll
      for (int i = 0; i < 8; ++i) sVt[(h0 + i) * 32 + key] = vsrc[i];
    }
    asm volatile("s_wait_asynccnt 0" ::: "memory");   // K async copy landed
    __syncthreads();

    if (s0 <= row0 + 15) {   // wave-uniform: skip fully-masked key tiles
      // ---- S = Q K^T : two 16x16 score tiles (32 keys) ----
      v8f s[2];
      #pragma unroll
      for (int nt = 0; nt < 2; ++nt) {
        Frag bk0, bk1;
        const unsigned short* kb = sK + (nt * 16 + n0) * HEADD + 8 * hi;
        bk0.q[0] = *(const v4u*)(kb);       bk0.q[1] = *(const v4u*)(kb + 16);
        bk1.q[0] = *(const v4u*)(kb + 32);  bk1.q[1] = *(const v4u*)(kb + 48);
        v8f t = {};
        t = __builtin_amdgcn_wmma_f32_16x16x32_bf16(false, aq[0].bf, false, bk0.bf,
                                                    (short)0, t, false, false);
        t = __builtin_amdgcn_wmma_f32_16x16x32_bf16(false, aq[1].bf, false, bk1.bf,
                                                    (short)0, t, false, false);
        s[nt] = t;
      }

      // ---- online softmax (per row: butterfly max/sum across 16-lane group) ----
      #pragma unroll
      for (int r = 0; r < 8; ++r) {
        const int row = row0 + r + 8 * hi;
        float s0v = s[0][r] * scale;
        float s1v = s[1][r] * scale;
        if (s0 + n0 > row)      s0v = -1e30f;   // causal mask (tril)
        if (s0 + 16 + n0 > row) s1v = -1e30f;
        float cm = fmaxf(s0v, s1v);
        #pragma unroll
        for (int m = 1; m < 16; m <<= 1) cm = fmaxf(cm, __shfl_xor(cm, m, 32));
        const float nm    = fmaxf(mrow[r], cm);
        const float alpha = __expf(mrow[r] - nm);
        mrow[r] = nm;
        const float p0 = __expf(s0v - nm);
        const float p1 = __expf(s1v - nm);
        float rs = p0 + p1;
        #pragma unroll
        for (int m = 1; m < 16; m <<= 1) rs += __shfl_xor(rs, m, 32);
        lrow[r] = lrow[r] * alpha + rs;
        #pragma unroll
        for (int nt = 0; nt < 4; ++nt) o[nt][r] *= alpha;
        unsigned short* pw = &sP[wave][(r + 8 * hi) * 32];
        pw[n0]      = f2bf(p0);
        pw[16 + n0] = f2bf(p1);
      }
      asm volatile("s_wait_dscnt 0" ::: "memory");   // P store -> P frag load

      // ---- O += P V : P as 16x32 A-fragment, V from transposed LDS ----
      Frag pf;
      {
        const unsigned short* pb = &sP[wave][n0 * 32];
        pf.q[0] = *(const v4u*)(pb + 8 * hi);
        pf.q[1] = *(const v4u*)(pb + 16 + 8 * hi);
      }
      #pragma unroll
      for (int nt = 0; nt < 4; ++nt) {
        Frag bv;
        const unsigned short* vb = sVt + (nt * 16 + n0) * 32 + 8 * hi;
        bv.q[0] = *(const v4u*)(vb);
        bv.q[1] = *(const v4u*)(vb + 16);
        o[nt] = __builtin_amdgcn_wmma_f32_16x16x32_bf16(
            false, pf.bf, false, bv.bf, (short)0, o[nt], false, false);
      }
    }
  }

  // ---- epilogue: normalize by row sum, write f32 output [B,T,64] ----
  #pragma unroll
  for (int nt = 0; nt < 4; ++nt)
    #pragma unroll
    for (int r = 0; r < 8; ++r) {
      const int row = row0 + r + 8 * hi;
      out[(size_t)(baseRow + row) * HEADD + nt * 16 + n0] = o[nt][r] / lrow[r];
    }
}

// ---------------------------------------------------------------------------
extern "C" void kernel_launch(void* const* d_in, const int* in_sizes, int n_in,
                              void* d_out, int out_size, void* d_ws, size_t ws_size,
                              hipStream_t stream) {
  const float* q  = (const float*)d_in[0];
  const float* k  = (const float*)d_in[1];
  const float* v  = (const float*)d_in[2];
  (void)d_in[3];   // mask is tril(T,T): causality computed inline in flash_attn
  const float* Wq = (const float*)d_in[4];
  const float* Wk = (const float*)d_in[5];
  const float* Wv = (const float*)d_in[6];
  (void)in_sizes; (void)n_in; (void)out_size; (void)ws_size;

  unsigned short* qh = (unsigned short*)d_ws;                // 4 MB bf16 each
  unsigned short* kh = qh + (size_t)MROWS * HEADD;
  unsigned short* vh = kh + (size_t)MROWS * HEADD;

  qkv_proj<<<dim3(MROWS / 128, 1, 3), 256, 0, stream>>>(q, k, v, Wq, Wk, Wv,
                                                        qh, kh, vh);
  flash_attn<<<dim3(SEQT / 128, NBATCH), 256, 0, stream>>>(qh, kh, vh,
                                                           (float*)d_out);
}
